// All4ConcatDecoder256_64_LongRange_5231270166859
// MI455X (gfx1250) — matrix operand order repaired and checked by reading, not compiled, not run on hardware
//
#include <hip/hip_runtime.h>
#include <hip/hip_bf16.h>
#include <stdint.h>

// ---------------------------------------------------------------------------
// CDNA5 (gfx1250) types
// ---------------------------------------------------------------------------
typedef __bf16 bf16_t;
typedef bf16_t v16bf __attribute__((ext_vector_type(16)));
typedef float  v8f   __attribute__((ext_vector_type(8)));
typedef unsigned int u32x4 __attribute__((ext_vector_type(4)));
typedef int          i32x8 __attribute__((ext_vector_type(8)));
typedef int          i32x4 __attribute__((ext_vector_type(4)));

#define BN_EPS     1e-5f
#define INTERP_EPS 1e-8f

// ---------------------------------------------------------------------------
// Pack W[M][K] f32 -> bf16 in WMMA A-fragment lane order:
//   Wp[(mt*nK32 + k32)*32 + lane][h]  (16 halfs contiguous per lane)
// Lane (lm,hi), half h: K offset = hi*16 + h.
// ---------------------------------------------------------------------------
__global__ __launch_bounds__(256)
void pack_w(const float* __restrict__ W, bf16_t* __restrict__ Wp, int M, int K) {
  long t = (long)blockIdx.x * 256 + threadIdx.x;
  if (t >= (long)M * K) return;
  const int nK32 = K >> 5;
  const int h    = (int)(t & 15);
  const int lane = (int)((t >> 4) & 31);
  const long tile = t >> 9;             // mt*nK32 + k32
  const int k32 = (int)(tile % nK32);
  const int mt  = (int)(tile / nK32);
  const int lm = lane & 15, hi = lane >> 4;
  const int v = h >> 1, p = h & 1;
  const int kb = hi * 16 + h;
  Wp[t] = (bf16_t)W[(size_t)(mt * 16 + lm) * K + k32 * 32 + kb];
}

// ---------------------------------------------------------------------------
// Transpose-convert X[b][C][N] f32 -> Xt[b][n][Crow] bf16 at column offset.
// ---------------------------------------------------------------------------
__global__ __launch_bounds__(256)
void xt_convert(const float* __restrict__ X, bf16_t* __restrict__ Xt,
                int C, int N, int Crow, int coff) {
  const int b = blockIdx.y;
  long t = (long)blockIdx.x * 256 + threadIdx.x;
  if (t >= (long)C * N) return;
  const int c = (int)(t % C);
  const int n = (int)(t / C);
  Xt[((size_t)b * N + n) * Crow + coff + c] =
      (bf16_t)X[((size_t)b * C + c) * N + n];
}

// ---------------------------------------------------------------------------
// WMMA GEMM: Y[b][m][n] = sum_k W[m][k] * X[b][k][n]
//   Wp : bf16, pre-packed A-fragment order
//   Xt : bf16, transposed [b][n][K]  (K multiple of 32)
// One wave computes a 16x64 tile: A loaded once per K-step (one 32B vector
// load), reused for 4 B tiles (one 32B vector load each) -> 4 v_wmma per step.
// ---------------------------------------------------------------------------
__global__ __launch_bounds__(32)
void wmma_gemm(const bf16_t* __restrict__ Wp, const bf16_t* __restrict__ Xt,
               float* __restrict__ Y, int M, int N, int K) {
  const int nK32 = K >> 5;
  const int b    = blockIdx.z;
  const int mt   = blockIdx.y;           // 16-row tile
  const int n0   = blockIdx.x * 64;      // 64-col supertile
  const int lane = threadIdx.x;
  const int lm   = lane & 15;
  const int hi   = lane >> 4;

  const bf16_t* Xb = Xt + (size_t)b * N * K;
  float*        Yb = Y + (size_t)b * M * N;

  v8f acc[4] = {};
  for (int k32 = 0; k32 < nK32; ++k32) {
    // A: one aligned 32-byte load in fragment order
    const v16bf a =
        *(const v16bf*)(Wp + ((size_t)mt * nK32 + k32) * 512 + lane * 16);
    if (k32 + 1 < nK32)
      __builtin_prefetch(Wp + ((size_t)mt * nK32 + k32 + 1) * 512 + lane * 16,
                         0, 3);
#pragma unroll
    for (int t = 0; t < 4; ++t) {
      const int n = n0 + t * 16 + lm;
      // B: lane holds K = k32*32 + hi*16 + h -> 16 contiguous halfs
      const v16bf bb = *(const v16bf*)(Xb + (size_t)n * K + k32 * 32 + hi * 16);
      acc[t] = __builtin_amdgcn_wmma_f32_16x16x32_bf16(
          false, a, false, bb, (short)0, acc[t], false, false);
    }
  }
#pragma unroll
  for (int t = 0; t < 4; ++t)
#pragma unroll
    for (int r = 0; r < 8; ++r)
      Yb[(size_t)(mt * 16 + r + hi * 8) * N + (n0 + t * 16 + lm)] = acc[t][r];
}

// ---------------------------------------------------------------------------
// BatchNorm stats: one block per channel, reduce over (B, N); fused scale/shift.
// ---------------------------------------------------------------------------
__global__ __launch_bounds__(256)
void bn_stats(const float* __restrict__ Y, int Cout, int N, int B,
              const float* __restrict__ g, const float* __restrict__ beta,
              float* __restrict__ scale, float* __restrict__ shift) {
  const int ch = blockIdx.x;
  __shared__ float ssum[256];
  __shared__ float ssq[256];
  float s = 0.f, q = 0.f;
  for (int b = 0; b < B; ++b) {
    const float* row = Y + ((size_t)b * Cout + ch) * N;
    for (int n = threadIdx.x; n < N; n += 256) {
      float v = row[n];
      s += v;
      q += v * v;
    }
  }
  ssum[threadIdx.x] = s;
  ssq[threadIdx.x]  = q;
  __syncthreads();
  for (int off = 128; off > 0; off >>= 1) {
    if ((int)threadIdx.x < off) {
      ssum[threadIdx.x] += ssum[threadIdx.x + off];
      ssq[threadIdx.x]  += ssq[threadIdx.x + off];
    }
    __syncthreads();
  }
  if (threadIdx.x == 0) {
    const float cnt = (float)B * (float)N;
    const float m   = ssum[0] / cnt;
    const float var = ssq[0] / cnt - m * m;
    const float sc  = g[ch] * rsqrtf(var + BN_EPS);
    scale[ch] = sc;
    shift[ch] = beta[ch] - m * sc;
  }
}

// In-place f32 normalize + ReLU (for tensors consumed by gathers / output).
__global__ __launch_bounds__(256)
void bn_apply_relu(float* __restrict__ Y, const float* __restrict__ scale,
                   const float* __restrict__ shift, int Cout, int N, long total) {
  long i = (long)blockIdx.x * 256 + threadIdx.x;
  if (i >= total) return;
  const int ch = (int)((i / N) % Cout);
  float v = fmaf(Y[i], scale[ch], shift[ch]);
  Y[i] = v > 0.f ? v : 0.f;
}

// Normalize + ReLU straight into a transposed bf16 GEMM operand.
__global__ __launch_bounds__(256)
void bn_apply_relu_t(const float* __restrict__ Y, const float* __restrict__ scale,
                     const float* __restrict__ shift, bf16_t* __restrict__ Xt,
                     int Cout, int N, int Crow, int coff, long total) {
  long i = (long)blockIdx.x * 256 + threadIdx.x;
  if (i >= total) return;
  const int n = (int)(i % N);
  const int c = (int)((i / N) % Cout);
  const int b = (int)(i / ((long)N * Cout));
  float v = fmaf(Y[i], scale[c], shift[c]);
  v = v > 0.f ? v : 0.f;
  Xt[((size_t)b * N + n) * Crow + coff + c] = (bf16_t)v;
}

// ---------------------------------------------------------------------------
// Brute-force 3-NN, source cloud staged in LDS.  The LDS fill is kicked off
// with the gfx1250 Tensor Data Mover (1-D tile of Ns*3 dwords into dynamic-LDS
// offset 0, D# per ISA ch.8), then a cooperative copy writes the identical
// data as a safety net before the barrier.
// ---------------------------------------------------------------------------
__global__ __launch_bounds__(256)
void knn3(const float* __restrict__ Pt, const float* __restrict__ Ps,
          int Nt, int Ns, int* __restrict__ idx, float* __restrict__ wgt) {
  extern __shared__ float sp[];            // Ns * 3 floats at LDS offset 0
  const int b = blockIdx.y;
  const float* ps = Ps + (size_t)b * Ns * 3;

#if defined(__gfx1250__) && __has_builtin(__builtin_amdgcn_tensor_load_to_lds) && \
    __has_builtin(__builtin_amdgcn_s_wait_tensorcnt)
  if (threadIdx.x < 32) {                  // wave 0 issues the TDM op
    const unsigned long long ga = (unsigned long long)(uintptr_t)ps;
    const unsigned ne = (unsigned)(Ns * 3);          // elements, data_size=4B
    u32x4 g0;
    g0[0] = 1u;                                      // count=1 user descriptor
    g0[1] = 0u;                                      // lds_addr = 0
    g0[2] = (unsigned)(ga & 0xFFFFFFFFu);            // global_addr[31:0]
    g0[3] = (unsigned)((ga >> 32) & 0x01FFFFFFu) | (2u << 30);  // addr hi|type=2
    i32x8 g1;
    g1[0] = (int)(2u << 16);                         // data_size = 4 bytes
    g1[1] = (int)((ne & 0xFFFFu) << 16);             // tensor_dim0[15:0]
    g1[2] = (int)((ne >> 16) | (1u << 16));          // dim0[31:16] | tensor_dim1=1
    g1[3] = (int)((ne & 0xFFFFu) << 16);             // tile_dim0 = ne
    g1[4] = 1;                                       // tile_dim1 = 1
    g1[5] = (int)ne;                                 // tensor_dim0_stride
    g1[6] = 0;
    g1[7] = 0;
    const i32x4 gz4 = {0, 0, 0, 0};
    const i32x8 gz8 = {0, 0, 0, 0, 0, 0, 0, 0};
    __builtin_amdgcn_tensor_load_to_lds(g0, g1, gz4, gz4, gz8, 0);
    __builtin_amdgcn_s_wait_tensorcnt(0);
  }
  __syncthreads();
#endif
  // Safety-net cooperative fill (identical data).
  for (int j = threadIdx.x; j < Ns * 3; j += 256) sp[j] = ps[j];
  __syncthreads();

  const int t = blockIdx.x * 256 + threadIdx.x;
  if (t >= Nt) return;
  const float* pt = Pt + ((size_t)b * Nt + t) * 3;
  const float px = pt[0], py = pt[1], pz = pt[2];

  float d0 = 1e30f, d1 = 1e30f, d2 = 1e30f;
  int   i0 = 0, i1 = 0, i2 = 0;
  for (int j = 0; j < Ns; ++j) {
    const float dx = px - sp[3 * j + 0];
    const float dy = py - sp[3 * j + 1];
    const float dz = pz - sp[3 * j + 2];
    const float d  = dx * dx + dy * dy + dz * dz;
    if (d < d0)      { d2 = d1; i2 = i1; d1 = d0; i1 = i0; d0 = d; i0 = j; }
    else if (d < d1) { d2 = d1; i2 = i1; d1 = d;  i1 = j; }
    else if (d < d2) { d2 = d;  i2 = j; }
  }
  const float r0 = 1.f / (d0 + INTERP_EPS);
  const float r1 = 1.f / (d1 + INTERP_EPS);
  const float r2 = 1.f / (d2 + INTERP_EPS);
  const float rs = 1.f / (r0 + r1 + r2);
  const size_t o = ((size_t)b * Nt + t) * 3;
  idx[o] = i0; idx[o + 1] = i1; idx[o + 2] = i2;
  wgt[o] = r0 * rs; wgt[o + 1] = r1 * rs; wgt[o + 2] = r2 * rs;
}

// ---------------------------------------------------------------------------
// 3-NN weighted gather writing straight into a transposed bf16 GEMM operand:
//   Out[b][n][coff + c] = bf16( sum_k w_k * H[b][c][idx_k] )
// c fastest across threads -> contiguous bf16 writes, idx/wgt broadcast.
// ---------------------------------------------------------------------------
__global__ __launch_bounds__(256)
void interp_gather_t(const float* __restrict__ H, int C, int Nsrc,
                     const int* __restrict__ idx, const float* __restrict__ wgt,
                     int Nt, bf16_t* __restrict__ Out, int Crow, int coff) {
  const int b = blockIdx.y;
  long tid = (long)blockIdx.x * 256 + threadIdx.x;
  if (tid >= (long)C * Nt) return;
  const int c = (int)(tid % C);
  const int n = (int)(tid / C);
  const size_t o = ((size_t)b * Nt + n) * 3;
  const int i0 = idx[o], i1 = idx[o + 1], i2 = idx[o + 2];
  const float w0 = wgt[o], w1 = wgt[o + 1], w2 = wgt[o + 2];
  const float* h = H + ((size_t)b * C + c) * Nsrc;
  Out[((size_t)b * Nt + n) * Crow + coff + c] =
      (bf16_t)(w0 * h[i0] + w1 * h[i1] + w2 * h[i2]);
}

// ---------------------------------------------------------------------------
// Host side
// ---------------------------------------------------------------------------
static inline size_t alignup(size_t x) { return (x + 255) & ~(size_t)255; }

extern "C" void kernel_launch(void* const* d_in, const int* in_sizes, int n_in,
                              void* d_out, int out_size, void* d_ws, size_t ws_size,
                              hipStream_t stream) {
  (void)in_sizes; (void)n_in; (void)out_size; (void)ws_size;
  const int B = 4;
  const int N1 = 16384, N2 = 4096, N3 = 1024, N4 = 256;

  const float* p1 = (const float*)d_in[0];
  const float* p2 = (const float*)d_in[1];
  const float* p3 = (const float*)d_in[2];
  const float* p4 = (const float*)d_in[3];
  const float* f1 = (const float*)d_in[4];   // [B,64,N1]
  const float* f2 = (const float*)d_in[5];   // [B,128,N2]
  const float* f3 = (const float*)d_in[6];   // [B,256,N3]
  const float* f4 = (const float*)d_in[7];   // [B,512,N4]
  const float* t4W = (const float*)d_in[8];
  const float* t3W = (const float*)d_in[9];
  const float* t2W = (const float*)d_in[10];
  const float* fuW = (const float*)d_in[11];
  const float* p1W = (const float*)d_in[12];
  const float* p2W = (const float*)d_in[13];
  const float* t4g = (const float*)d_in[14]; const float* t4b = (const float*)d_in[15];
  const float* t3g = (const float*)d_in[16]; const float* t3b = (const float*)d_in[17];
  const float* t2g = (const float*)d_in[18]; const float* t2b = (const float*)d_in[19];
  const float* fug = (const float*)d_in[20]; const float* fub = (const float*)d_in[21];
  const float* p1g = (const float*)d_in[22]; const float* p1b = (const float*)d_in[23];
  const float* p2g = (const float*)d_in[24]; const float* p2b = (const float*)d_in[25];

  // ---- workspace carve-out ------------------------------------------------
  char* ws = (char*)d_ws;
  size_t off = 0;
  auto alloc = [&](size_t bytes) { char* p = ws + off; off += alignup(bytes); return p; };

  // packed bf16 weights (A-fragment order)
  bf16_t* Wp4  = (bf16_t*)alloc((size_t)256 * 512 * 2);
  bf16_t* Wp3  = (bf16_t*)alloc((size_t)256 * 256 * 2);
  bf16_t* Wp2  = (bf16_t*)alloc((size_t)128 * 128 * 2);
  bf16_t* Wpf  = (bf16_t*)alloc((size_t)128 * 640 * 2);
  bf16_t* Wpa  = (bf16_t*)alloc((size_t)64  * 192 * 2);
  bf16_t* Wpb  = (bf16_t*)alloc((size_t)64  * 64  * 2);
  // transposed bf16 activations
  bf16_t* Xt4    = (bf16_t*)alloc((size_t)B * N4 * 512 * 2);
  bf16_t* Xt3    = (bf16_t*)alloc((size_t)B * N3 * 256 * 2);
  bf16_t* Xt2    = (bf16_t*)alloc((size_t)B * N2 * 128 * 2);
  bf16_t* cat640 = (bf16_t*)alloc((size_t)B * N2 * 640 * 2);  // [f2t;f3t;f4t]^T
  bf16_t* cat192 = (bf16_t*)alloc((size_t)B * N1 * 192 * 2);  // [f1;fi]^T
  bf16_t* fp1t   = (bf16_t*)alloc((size_t)B * N1 * 64  * 2);
  // f32 conv outputs (pre/post BN)
  float* h4    = (float*)alloc((size_t)B * 256 * N4 * 4);
  float* h3    = (float*)alloc((size_t)B * 256 * N3 * 4);
  float* h2    = (float*)alloc((size_t)B * 128 * N2 * 4);
  float* fuse  = (float*)alloc((size_t)B * 128 * N2 * 4);
  float* fp1y  = (float*)alloc((size_t)B * 64  * N1 * 4);
  // kNN results
  int*   i24 = (int*)  alloc((size_t)B * N2 * 3 * 4);
  float* w24 = (float*)alloc((size_t)B * N2 * 3 * 4);
  int*   i23 = (int*)  alloc((size_t)B * N2 * 3 * 4);
  float* w23 = (float*)alloc((size_t)B * N2 * 3 * 4);
  int*   i22 = (int*)  alloc((size_t)B * N2 * 3 * 4);
  float* w22 = (float*)alloc((size_t)B * N2 * 3 * 4);
  int*   i12 = (int*)  alloc((size_t)B * N1 * 3 * 4);
  float* w12 = (float*)alloc((size_t)B * N1 * 3 * 4);
  float* sc  = (float*)alloc(6 * 256 * 4);
  float* sh  = (float*)alloc(6 * 256 * 4);

  float* out = (float*)d_out;   // [B,64,N1]

  auto cdiv = [](long a, long b) { return (unsigned)((a + b - 1) / b); };

  // ---- weight packing -----------------------------------------------------
  pack_w<<<cdiv((long)256 * 512, 256), 256, 0, stream>>>(t4W, Wp4, 256, 512);
  pack_w<<<cdiv((long)256 * 256, 256), 256, 0, stream>>>(t3W, Wp3, 256, 256);
  pack_w<<<cdiv((long)128 * 128, 256), 256, 0, stream>>>(t2W, Wp2, 128, 128);
  pack_w<<<cdiv((long)128 * 640, 256), 256, 0, stream>>>(fuW, Wpf, 128, 640);
  pack_w<<<cdiv((long)64  * 192, 256), 256, 0, stream>>>(p1W, Wpa, 64, 192);
  pack_w<<<cdiv((long)64  * 64 , 256), 256, 0, stream>>>(p2W, Wpb, 64, 64);

  // ---- activation transpose/convert to bf16 -------------------------------
  xt_convert<<<dim3(cdiv((long)512 * N4, 256), B), 256, 0, stream>>>(f4, Xt4, 512, N4, 512, 0);
  xt_convert<<<dim3(cdiv((long)256 * N3, 256), B), 256, 0, stream>>>(f3, Xt3, 256, N3, 256, 0);
  xt_convert<<<dim3(cdiv((long)128 * N2, 256), B), 256, 0, stream>>>(f2, Xt2, 128, N2, 128, 0);
  xt_convert<<<dim3(cdiv((long)64  * N1, 256), B), 256, 0, stream>>>(f1, cat192, 64, N1, 192, 0);

  // ---- 3-NN searches ------------------------------------------------------
  {
    dim3 g24((N2 + 255) / 256, B);
    knn3<<<g24, 256, (size_t)N4 * 3 * 4, stream>>>(p2, p4, N2, N4, i24, w24);
    knn3<<<g24, 256, (size_t)N3 * 3 * 4, stream>>>(p2, p3, N2, N3, i23, w23);
    knn3<<<g24, 256, (size_t)N2 * 3 * 4, stream>>>(p2, p2, N2, N2, i22, w22);
    dim3 g12((N1 + 255) / 256, B);
    knn3<<<g12, 256, (size_t)N2 * 3 * 4, stream>>>(p1, p2, N1, N2, i12, w12);
  }

  // conv = WMMA GEMM + BN stats + in-place f32 apply
  auto conv = [&](const bf16_t* Wp, const bf16_t* Xt, float* Y, int M, int N,
                  int K, const float* g, const float* bt, int slot) {
    dim3 grid(N / 64, M / 16, B);
    wmma_gemm<<<grid, 32, 0, stream>>>(Wp, Xt, Y, M, N, K);
    bn_stats<<<M, 256, 0, stream>>>(Y, M, N, B, g, bt, sc + slot * 256, sh + slot * 256);
    long total = (long)B * M * N;
    bn_apply_relu<<<cdiv(total, 256), 256, 0, stream>>>(
        Y, sc + slot * 256, sh + slot * 256, M, N, total);
  };

  // ---- transform convs ----------------------------------------------------
  conv(Wp4, Xt4, h4, 256, N4, 512, t4g, t4b, 0);
  conv(Wp3, Xt3, h3, 256, N3, 256, t3g, t3b, 1);
  conv(Wp2, Xt2, h2, 128, N2, 128, t2g, t2b, 2);

  // ---- interpolate each scale onto p2, straight into bf16-T concat --------
  interp_gather_t<<<dim3(cdiv((long)128 * N2, 256), B), 256, 0, stream>>>(
      h2, 128, N2, i22, w22, N2, cat640, 640, 0);
  interp_gather_t<<<dim3(cdiv((long)256 * N2, 256), B), 256, 0, stream>>>(
      h3, 256, N3, i23, w23, N2, cat640, 640, 128);
  interp_gather_t<<<dim3(cdiv((long)256 * N2, 256), B), 256, 0, stream>>>(
      h4, 256, N4, i24, w24, N2, cat640, 640, 384);

  // ---- fuse conv over 640-channel concat ----------------------------------
  conv(Wpf, cat640, fuse, 128, N2, 640, fug, fub, 3);

  // ---- upsample fuse to p1, straight into bf16-T concat cols 64..191 ------
  interp_gather_t<<<dim3(cdiv((long)128 * N1, 256), B), 256, 0, stream>>>(
      fuse, 128, N2, i12, w12, N1, cat192, 192, 64);

  // ---- fp1: GEMM + stats, apply straight into bf16-T operand for fp2 ------
  {
    dim3 grid(N1 / 64, 64 / 16, B);
    wmma_gemm<<<grid, 32, 0, stream>>>(Wpa, cat192, fp1y, 64, N1, 192);
    bn_stats<<<64, 256, 0, stream>>>(fp1y, 64, N1, B, p1g, p1b, sc + 4 * 256, sh + 4 * 256);
    long total = (long)B * 64 * N1;
    bn_apply_relu_t<<<cdiv(total, 256), 256, 0, stream>>>(
        fp1y, sc + 4 * 256, sh + 4 * 256, fp1t, 64, N1, 64, 0, total);
  }

  // ---- fp2 conv, result straight into d_out -------------------------------
  conv(Wpb, fp1t, out, 64, N1, 64, p2g, p2b, 5);
}
